// BidirectionalSelfAttention_20761871908912
// MI455X (gfx1250) — matrix operand order repaired and checked by reading, compiled, not run on hardware
//
#include <hip/hip_runtime.h>

// ---------------------------------------------------------------------------
// Types for CDNA5 WMMA (wave32): 16x16x32 bf16 -> f32
// ---------------------------------------------------------------------------
typedef __attribute__((ext_vector_type(16))) __bf16 v16bf;
typedef __attribute__((ext_vector_type(8)))  float  v8f;
typedef int v4i __attribute__((vector_size(16)));  // matches async-LDS builtin param

union FragAB {
  v16bf v;
  uint4 q[2];
  unsigned short us[16];
};

__device__ __forceinline__ unsigned short f2bf(float f) {
  union { float f; unsigned int u; } v; v.f = f;
  unsigned int r = v.u + 0x7FFFu + ((v.u >> 16) & 1u);  // round-to-nearest-even
  return (unsigned short)(r >> 16);
}
__device__ __forceinline__ unsigned int pack2(float lo, float hi) {
  return (unsigned int)f2bf(lo) | ((unsigned int)f2bf(hi) << 16);
}
__device__ __forceinline__ float bf2f(unsigned short h) {
  union { unsigned int u; float f; } v; v.u = ((unsigned int)h) << 16;
  return v.f;
}
__device__ __forceinline__ v8f zero8() {
  v8f z;
#pragma unroll
  for (int i = 0; i < 8; i++) z[i] = 0.0f;
  return z;
}

// ---------------------------------------------------------------------------
// CDNA5 async global->LDS copy (ASYNCcnt path) with safe fallback.
// Signature confirmed: (int4 AS1*, int4 AS3*, imm int offset, imm int cpol)
// ---------------------------------------------------------------------------
#if defined(__has_builtin)
#if __has_builtin(__builtin_amdgcn_global_load_async_to_lds_b128)
#define HAS_ASYNC_LDS 1
#endif
#endif

typedef __attribute__((address_space(1))) v4i* gv4i_p;
typedef __attribute__((address_space(3))) v4i* lv4i_p;

__device__ __forceinline__ void async_copy_b128(const unsigned short* g,
                                                unsigned short* l) {
#ifdef HAS_ASYNC_LDS
  __builtin_amdgcn_global_load_async_to_lds_b128(
      (gv4i_p)(unsigned short*)g, (lv4i_p)l, 0, 0);
#else
  *(uint4*)l = *(const uint4*)g;
#endif
}

__device__ __forceinline__ void wait_async_all() {
#ifdef HAS_ASYNC_LDS
#if __has_builtin(__builtin_amdgcn_s_wait_asynccnt)
  __builtin_amdgcn_s_wait_asynccnt(0);
#else
  asm volatile("s_wait_asynccnt 0x0" ::: "memory");
#endif
#endif
}

// ---------------------------------------------------------------------------
// Fragment loaders (wave32 layouts per CDNA5 ISA 7.12.2)
// ---------------------------------------------------------------------------
// A-operand frag (16 rows x 32 K), source row-major [m][k], lda multiple of 8.
__device__ __forceinline__ v16bf load_frag_a(const unsigned short* base, int lda) {
  int lane = threadIdx.x & 31;
  int m = lane & 15;
  int kb = (lane >> 4) << 3;  // 0 or 8
  const unsigned short* p = base + (size_t)m * lda + kb;
  FragAB f;
  f.q[0] = *(const uint4*)(p);
  f.q[1] = *(const uint4*)(p + 16);
  return f.v;
}

// B-operand frag (32 K x 16 cols), source stored TRANSPOSED [n][k], lda mult of 8.
__device__ __forceinline__ v16bf load_frag_b(const unsigned short* base, int lda) {
  int lane = threadIdx.x & 31;
  int n = lane & 15;
  int kb = (lane >> 4) << 3;  // 0 or 8
  const unsigned short* p = base + (size_t)n * lda + kb;
  FragAB f;
  f.q[0] = *(const uint4*)(p);
  f.q[1] = *(const uint4*)(p + 16);
  return f.v;
}

__device__ __forceinline__ v8f wmma_bf16(v16bf a, v16bf b, v8f c) {
  return __builtin_amdgcn_wmma_f32_16x16x32_bf16(
      /*neg_a=*/false, a, /*neg_b=*/false, b,
      /*c_mod=*/(short)0, c, /*reuse_a=*/false, /*reuse_b=*/false);
}

// ---------------------------------------------------------------------------
// GEMM: Out[M,N] = A[M,K] @ B[K,N].  Block tile 256x256, K-step 32,
// 512 threads = 16 waves in a 4x4 grid; each wave owns a 64x64 sub-tile
// (4x4 WMMA accumulators).  256x256 tiling halves L2 re-read traffic vs
// 128x128 (A read N/256 times, B read M/256 times).
// A: f32 (pack-convert) or bf16 (async b128 copy).  B: f32, transposed into
// LDS [n][k].  All LDS traffic is b128.
// ---------------------------------------------------------------------------
template <bool A_F32, bool OUT_F32>
__global__ __launch_bounds__(512) void gemm_bf16_kernel(
    const void* __restrict__ Aptr, const float* __restrict__ B,
    void* __restrict__ Out, int M, int N, int K) {
  __shared__ unsigned short As[256 * 40];  // [m][k], pad stride 40  (20 KB)
  __shared__ unsigned short Bs[256 * 40];  // [n][k], pad stride 40  (20 KB)
  (void)M;

  const int tid = threadIdx.x;
  const int w = tid >> 5;
  const int wm = w & 3;   // 64-row group
  const int wn = w >> 2;  // 64-col group
  const int bm = blockIdx.y * 256;
  const int bn = blockIdx.x * 256;

  v8f acc[4][4];
#pragma unroll
  for (int i = 0; i < 4; i++)
#pragma unroll
    for (int j = 0; j < 4; j++) acc[i][j] = zero8();

  for (int k0 = 0; k0 < K; k0 += 32) {
    __syncthreads();
    // ---- stage A tile 256x32 (1024 x 8-element tasks) ----
#pragma unroll
    for (int i = 0; i < 2; i++) {
      int tt = tid + i * 512;
      int r = tt >> 2, kc = (tt & 3) * 8;
      if (A_F32) {
        const float* A = (const float*)Aptr;
        const float* g = &A[(size_t)(bm + r) * K + k0 + kc];
        float4 f0 = *(const float4*)g;
        float4 f1 = *(const float4*)(g + 4);
        uint4 pk = make_uint4(pack2(f0.x, f0.y), pack2(f0.z, f0.w),
                              pack2(f1.x, f1.y), pack2(f1.z, f1.w));
        *(uint4*)&As[r * 40 + kc] = pk;
      } else {
        const unsigned short* A = (const unsigned short*)Aptr;
        async_copy_b128(&A[(size_t)(bm + r) * K + k0 + kc], &As[r * 40 + kc]);
      }
    }
    // ---- stage B tile 32x256 transposed -> Bs[n][k] ----
#pragma unroll
    for (int i = 0; i < 2; i++) {
      int tt = tid + i * 512;
      int n = tt & 255, kc = (tt >> 8) * 8;
      unsigned int p0, p1, p2, p3;
      {
        const float* g = &B[(size_t)(k0 + kc) * N + bn + n];
        float a0 = g[0];
        float a1 = g[(size_t)N];
        float a2 = g[(size_t)2 * N];
        float a3 = g[(size_t)3 * N];
        float a4 = g[(size_t)4 * N];
        float a5 = g[(size_t)5 * N];
        float a6 = g[(size_t)6 * N];
        float a7 = g[(size_t)7 * N];
        p0 = pack2(a0, a1); p1 = pack2(a2, a3);
        p2 = pack2(a4, a5); p3 = pack2(a6, a7);
      }
      *(uint4*)&Bs[n * 40 + kc] = make_uint4(p0, p1, p2, p3);
    }
    // prefetch next K-step tiles while this one is consumed
    if (k0 + 32 < K) {
      int n = tid & 255, kc = (tid >> 8) * 8;
      __builtin_prefetch(&B[(size_t)(k0 + 32 + kc) * N + bn + n], 0, 1);
      int r = tid >> 2, kc2 = (tid & 3) * 8;
      if (A_F32)
        __builtin_prefetch(&((const float*)Aptr)[(size_t)(bm + r) * K + k0 + 32 + kc2], 0, 1);
    }
    if (!A_F32) wait_async_all();
    __syncthreads();

    v16bf afr[4], bfr[4];
#pragma unroll
    for (int i = 0; i < 4; i++)
      afr[i] = load_frag_a(&As[(wm * 64 + i * 16) * 40], 40);
#pragma unroll
    for (int j = 0; j < 4; j++)
      bfr[j] = load_frag_b(&Bs[(wn * 64 + j * 16) * 40], 40);
#pragma unroll
    for (int i = 0; i < 4; i++)
#pragma unroll
      for (int j = 0; j < 4; j++)
        acc[i][j] = wmma_bf16(afr[i], bfr[j], acc[i][j]);
  }

  // ---- epilogue: C/D layout -> memory ----
  const int lane = tid & 31;
  const int rg = (lane >> 4) << 3;
  const int cn = lane & 15;
#pragma unroll
  for (int i = 0; i < 4; i++)
#pragma unroll
    for (int j = 0; j < 4; j++)
#pragma unroll
      for (int r = 0; r < 8; r++) {
        int row = bm + wm * 64 + i * 16 + rg + r;
        int col = bn + wn * 64 + j * 16 + cn;
        float v = acc[i][j][r];
        if (OUT_F32)
          ((float*)Out)[(size_t)row * N + col] = v;
        else
          ((unsigned short*)Out)[(size_t)row * N + col] = f2bf(v);
      }
}

// ---------------------------------------------------------------------------
// RoPE (interleaved pairs), in-place on q,k slices of bf16 qkv buffer.
// ---------------------------------------------------------------------------
__global__ __launch_bounds__(256) void rope_kernel(unsigned short* __restrict__ qkv) {
  int p = blockIdx.x * 256 + threadIdx.x;  // pair index
  int i = p & 63;
  int h = (p >> 6) & 15;
  int t = (p >> 10) & 2047;
  int b = p >> 21;
  float inv_freq = __powf(10000.0f, -((float)(2 * i)) * (1.0f / 128.0f));
  float ang = (float)t * inv_freq;
  float c = __cosf(ang), s = __sinf(ang);
  size_t off = ((size_t)(b * 2048 + t)) * 6144 + h * 128 + 2 * i;
  {  // q
    float x0 = bf2f(qkv[off]), x1 = bf2f(qkv[off + 1]);
    qkv[off] = f2bf(x0 * c - x1 * s);
    qkv[off + 1] = f2bf(x1 * c + x0 * s);
  }
  {  // k
    size_t ok = off + 2048;
    float x0 = bf2f(qkv[ok]), x1 = bf2f(qkv[ok + 1]);
    qkv[ok] = f2bf(x0 * c - x1 * s);
    qkv[ok + 1] = f2bf(x1 * c + x0 * s);
  }
}

// ---------------------------------------------------------------------------
// Flash attention: grid (T/128, B*H), 8 waves; each wave owns 16 q-rows with
// full 128-d accumulators; K/V streamed in 64-token tiles (K via async LDS,
// V transposed with token-pair packed b32 stores).
// ---------------------------------------------------------------------------
__global__ __launch_bounds__(256) void attn_kernel(
    const unsigned short* __restrict__ qkv, unsigned short* __restrict__ y) {
  __shared__ unsigned short Ks[64 * 136];    // [token][d]            17,408 B
  __shared__ unsigned short Vs[128 * 72];    // [d][token] transposed 18,432 B
  __shared__ unsigned short Ps[8][16 * 72];  // per-wave P [m][k]     18,432 B

  const int tid = threadIdx.x;
  const int w = tid >> 5;
  const int lane = tid & 31;
  const int b = blockIdx.y >> 4;
  const int h = blockIdx.y & 15;
  const int qrow0 = blockIdx.x * 128 + w * 16;

  const int rowstride = 6144;
  const size_t hcol = (size_t)h * 128;

  // Preload Q fragments straight from global (A layout, 4 d-chunks of 32)
  v16bf qf[4];
  {
    const unsigned short* qp =
        qkv + ((size_t)(b * 2048 + qrow0)) * rowstride + hcol;
#pragma unroll
    for (int kk = 0; kk < 4; kk++) qf[kk] = load_frag_a(qp + kk * 32, rowstride);
  }

  v8f o[8];
#pragma unroll
  for (int i = 0; i < 8; i++) o[i] = zero8();
  float r_max[8], r_sum[8];
#pragma unroll
  for (int r = 0; r < 8; r++) { r_max[r] = -1e30f; r_sum[r] = 0.0f; }

  const float scale = 0.08838834764831845f;  // 1/sqrt(128)

  for (int kt = 0; kt < 2048; kt += 64) {
    __syncthreads();
    // ---- K: async b128 copies (1024 tasks of 8 elements) ----
#pragma unroll
    for (int i = 0; i < 4; i++) {
      int tt = tid + i * 256;
      int tok = tt >> 4, dc = (tt & 15) * 8;
      size_t grow = ((size_t)(b * 2048 + kt + tok)) * rowstride + hcol + dc;
      async_copy_b128(&qkv[grow + 2048], &Ks[tok * 136 + dc]);
    }
    // ---- V: token-pair loads, packed b32 transpose stores (512 tasks) ----
#pragma unroll
    for (int i = 0; i < 2; i++) {
      int tt = tid + i * 256;
      int tokp = tt >> 4, dc = (tt & 15) * 8;  // tokens 2*tokp, 2*tokp+1
      size_t grow =
          ((size_t)(b * 2048 + kt + 2 * tokp)) * rowstride + hcol + dc + 4096;
      uint4 v0 = *(const uint4*)&qkv[grow];
      uint4 v1 = *(const uint4*)&qkv[grow + rowstride];
      const unsigned short* a = (const unsigned short*)&v0;
      const unsigned short* c = (const unsigned short*)&v1;
#pragma unroll
      for (int j = 0; j < 8; j++) {
        unsigned int word = (unsigned int)a[j] | ((unsigned int)c[j] << 16);
        *(unsigned int*)&Vs[(dc + j) * 72 + 2 * tokp] = word;
      }
    }
    wait_async_all();
    __syncthreads();

    // ---- S = Q @ K^T : 4 token-tiles x 4 d-chunks ----
    v8f s[4];
#pragma unroll
    for (int nt = 0; nt < 4; nt++) {
      s[nt] = zero8();
#pragma unroll
      for (int kk = 0; kk < 4; kk++) {
        v16bf bfr = load_frag_b(&Ks[(nt * 16) * 136 + kk * 32], 136);
        s[nt] = wmma_bf16(qf[kk], bfr, s[nt]);
      }
    }

    // ---- online softmax (row stats across the 16-lane half-group) ----
#pragma unroll
    for (int r = 0; r < 8; r++) {
      float m0 = fmaxf(fmaxf(s[0][r], s[1][r]), fmaxf(s[2][r], s[3][r])) * scale;
#pragma unroll
      for (int md = 1; md < 16; md <<= 1) m0 = fmaxf(m0, __shfl_xor(m0, md, 32));
      float nm = fmaxf(r_max[r], m0);
      float corr = __expf(r_max[r] - nm);
      r_max[r] = nm;
      float p0 = __expf(s[0][r] * scale - nm);
      float p1 = __expf(s[1][r] * scale - nm);
      float p2 = __expf(s[2][r] * scale - nm);
      float p3 = __expf(s[3][r] * scale - nm);
      s[0][r] = p0; s[1][r] = p1; s[2][r] = p2; s[3][r] = p3;
      float ps = (p0 + p1) + (p2 + p3);
#pragma unroll
      for (int md = 1; md < 16; md <<= 1) ps += __shfl_xor(ps, md, 32);
      r_sum[r] = r_sum[r] * corr + ps;
#pragma unroll
      for (int dt = 0; dt < 8; dt++) o[dt][r] *= corr;
    }

    // ---- scatter P (C-layout) into per-wave LDS [m][k] ----
    {
      int rg = (lane >> 4) << 3, cn = lane & 15;
#pragma unroll
      for (int nt = 0; nt < 4; nt++)
#pragma unroll
        for (int r = 0; r < 8; r++)
          Ps[w][(rg + r) * 72 + nt * 16 + cn] = f2bf(s[nt][r]);
    }

    // ---- O += P @ V : 8 d-tiles x 2 token-chunks ----
    v16bf pf[2];
#pragma unroll
    for (int kk = 0; kk < 2; kk++) pf[kk] = load_frag_a(&Ps[w][kk * 32], 72);
#pragma unroll
    for (int dt = 0; dt < 8; dt++) {
#pragma unroll
      for (int kk = 0; kk < 2; kk++) {
        v16bf vfr = load_frag_b(&Vs[(dt * 16) * 72 + kk * 32], 72);
        o[dt] = wmma_bf16(pf[kk], vfr, o[dt]);
      }
    }
  }

  // ---- normalize and store y (bf16) ----
  {
    int rg = (lane >> 4) << 3, cn = lane & 15;
#pragma unroll
    for (int dt = 0; dt < 8; dt++)
#pragma unroll
      for (int r = 0; r < 8; r++) {
        int row = qrow0 + rg + r;
        float v = o[dt][r] / r_sum[r];
        y[((size_t)(b * 2048 + row)) * 2048 + hcol + dt * 16 + cn] = f2bf(v);
      }
  }
}

// ---------------------------------------------------------------------------
// Host-side launch
// ---------------------------------------------------------------------------
extern "C" void kernel_launch(void* const* d_in, const int* in_sizes, int n_in,
                              void* d_out, int out_size, void* d_ws, size_t ws_size,
                              hipStream_t stream) {
  (void)in_sizes; (void)n_in; (void)out_size; (void)ws_size;
  const float* x = (const float*)d_in[0];
  const float* w_qkv = (const float*)d_in[1];
  const float* w_proj = (const float*)d_in[2];
  float* out = (float*)d_out;

  unsigned short* qkv_ws = (unsigned short*)d_ws;       // 8192*6144 bf16 = 96 MiB
  unsigned short* y_ws = qkv_ws + (size_t)8192 * 6144;  // 8192*2048 bf16 = 32 MiB

  // 1) qkv = x @ w_qkv  (f32 -> bf16), 256x256 tiles
  gemm_bf16_kernel<true, false><<<dim3(24, 32), 512, 0, stream>>>(
      x, w_qkv, qkv_ws, 8192, 6144, 2048);
  // 2) RoPE in-place on q,k
  rope_kernel<<<dim3(8388608 / 256), 256, 0, stream>>>(qkv_ws);
  // 3) flash attention -> y (bf16)
  attn_kernel<<<dim3(16, 64), 256, 0, stream>>>(qkv_ws, y_ws);
  // 4) out = y @ w_proj  (bf16 A, f32 out), 256x256 tiles
  gemm_bf16_kernel<false, true><<<dim3(8, 32), 512, 0, stream>>>(
      y_ws, w_proj, out, 8192, 2048, 2048);
}
